// CDAttnBlock_12532714570347
// MI455X (gfx1250) — matrix operand order repaired and checked by reading, not compiled, not run on hardware
//
#include <hip/hip_runtime.h>
#include <hip/hip_bf16.h>
#include <cstdint>

// ---------------------------------------------------------------------------
// CDNA5 (gfx1250) fused dual/cross attention block, bf16 WMMA everywhere.
//   QKV = [x;x2] @ Wqkv     (bf16 WMMA GEMM; Q,K row-major, V transposed out)
//   o{1,2,3} = flashattn    (bf16 WMMA QK^T + PV, online softmax)
//   y{1,2,3} = o @ Wout + b (bf16 WMMA GEMM, f32 out)
// Weights are pre-transposed so B-side WMMA fragments are contiguous b128
// LDS loads; tile staging uses CDNA5 async global->LDS DMA (ASYNCcnt) with
// LDS double buffering to overlap DMA with WMMA math.
// ---------------------------------------------------------------------------

typedef __attribute__((ext_vector_type(16))) __bf16 v16bf;
typedef __attribute__((ext_vector_type(8)))  float  v8f;

union AFrag { v16bf v; uint32_t u[8]; };

__device__ __forceinline__ uint16_t f2bf(float f) {
  union { float f; uint32_t u; } c; c.f = f;
  uint32_t u = c.u;
  return (uint16_t)((u + 0x7FFFu + ((u >> 16) & 1u)) >> 16);  // RNE
}
// 16-bit A-matrix 16x32 K index for vector element pair {2j,2j+1}: K = 16*half + 2j (contiguous, matches B)
__device__ __forceinline__ int a_k(int j, int half) {
  return 16 * half + 2 * j;
}
// 16-bit B-matrix 32x16: lane group half holds K=16*half..16*half+15,
// element pair {2j,2j+1} -> K = 16*half + 2j (contiguous per lane).

// CDNA5 async global->LDS 128-bit DMA (tracked with ASYNCcnt).
__device__ __forceinline__ void async_b128(uint32_t lds_addr, uint64_t gaddr) {
  asm volatile("global_load_async_to_lds_b128 %0, %1, off"
               :: "v"(lds_addr), "v"(gaddr) : "memory");
}
__device__ __forceinline__ void wait_async0() {
  asm volatile("s_wait_asynccnt 0" ::: "memory");
}
__device__ __forceinline__ uint32_t lds_off(const void* p) {
  return (uint32_t)(uintptr_t)p;   // low 32 bits of flat addr == LDS offset
}

// ---------------------------------------------------------------------------
// bf16 WMMA GEMM: C[M,N] = A[M,K] @ BT[N,K]^T  (+bias, f32 out) or bf16 out.
// 128x128 block tile, 256 threads = 8 waves, wave tile 64x32 (4x2 frags).
// Both LDS tiles have identical geometry (128 rows x 32 k, 80B rows) and are
// filled by async DMA with double buffering. When Vt!=null, the V section of
// the QKV output (cols >= 1536) is scattered transposed as Vt[in][b][h][d][s].
// ---------------------------------------------------------------------------
__global__ __launch_bounds__(256) void wmma_gemm_bf16(
    const uint16_t* __restrict__ A, int lda, long long strideAz,
    const uint16_t* __restrict__ BT, int ldbt,
    uint16_t* __restrict__ Cbf, float* __restrict__ Cf, long long strideCz,
    int ldc, const float* __restrict__ bias, uint16_t* __restrict__ Vt,
    int M, int N, int K) {
  // 80B row stride: 16B-aligned for async b128, bank rotation 20 (conflict-free)
  __shared__ uint16_t As[2][128][40];    // [buf][m][k]
  __shared__ uint16_t Bts[2][128][40];   // [buf][n][k]

  const int tid  = threadIdx.x;
  const int lane = tid & 31, wid = tid >> 5;
  const int half = lane >> 4, ln = lane & 15;
  const int gn0 = blockIdx.x * 128, gm0 = blockIdx.y * 128;
  const int wm = (wid >> 2) * 64, wn = (wid & 3) * 32;
  A += (long long)blockIdx.z * strideAz;

  // per-thread staging geometry: 128 rows x 2 chunks(16B) per tile
  const int srow = tid >> 1, sc0 = (tid & 1) * 16;

  auto stage = [&](int buf, int kk) {
    const uint16_t* asrc = A  + (long long)(gm0 + srow) * lda  + kk + sc0;
    const uint16_t* bsrc = BT + (long long)(gn0 + srow) * ldbt + kk + sc0;
    uint32_t adst = lds_off(&As[buf][srow][sc0]);
    uint32_t bdst = lds_off(&Bts[buf][srow][sc0]);
    async_b128(adst,      (uint64_t)(uintptr_t)asrc);
    async_b128(adst + 16, (uint64_t)(uintptr_t)(asrc + 8));
    async_b128(bdst,      (uint64_t)(uintptr_t)bsrc);
    async_b128(bdst + 16, (uint64_t)(uintptr_t)(bsrc + 8));
  };

  v8f acc[4][2] = {};

  stage(0, 0);
  wait_async0();
  __syncthreads();

  int buf = 0;
  for (int kk = 0; kk < K; kk += 32) {
    if (kk + 32 < K) stage(buf ^ 1, kk + 32);   // overlap DMA with math

    AFrag af[4];
#pragma unroll
    for (int i = 0; i < 4; ++i) {
      int m = wm + 16 * i + ln;
#pragma unroll
      for (int j = 0; j < 8; ++j)
        af[i].u[j] = *(const uint32_t*)&As[buf][m][a_k(j, half)];
    }
    AFrag bfr[2];
#pragma unroll
    for (int jf = 0; jf < 2; ++jf) {
      int n = wn + 16 * jf + ln;
#pragma unroll
      for (int j = 0; j < 8; ++j)
        bfr[jf].u[j] = *(const uint32_t*)&Bts[buf][n][16 * half + 2 * j];
    }
#pragma unroll
    for (int i = 0; i < 4; ++i)
#pragma unroll
      for (int jf = 0; jf < 2; ++jf)
        acc[i][jf] = __builtin_amdgcn_wmma_f32_16x16x32_bf16(
            false, af[i].v, false, bfr[jf].v, (short)0, acc[i][jf], false, false);

    wait_async0();      // next tile resident
    __syncthreads();    // everyone done reading current tile
    buf ^= 1;
  }

  // epilogue: C layout VGPR e -> M = e + 8*half, N = lane&15
  const bool vpath = (Vt != nullptr) && (gn0 >= 1536);  // V section, uniform per block
#pragma unroll
  for (int i = 0; i < 4; ++i) {
#pragma unroll
    for (int jf = 0; jf < 2; ++jf) {
#pragma unroll
      for (int e = 0; e < 8; ++e) {
        int row = gm0 + wm + 16 * i + e + 8 * half;
        int col = gn0 + wn + 16 * jf + ln;
        float v = acc[i][jf][e];
        if (Cf) {
          Cf[(long long)blockIdx.z * strideCz + (long long)row * ldc + col] = v + bias[col];
        } else if (vpath) {
          int in = row >> 13, bb = (row >> 10) & 7, s = row & 1023;
          int hc = (col - 1536) >> 6, d = (col - 1536) & 63;
          Vt[((((long long)in * 8 + bb) * 12 + hc) * 64 + d) * 1024 + s] = f2bf(v);
        } else {
          Cbf[(long long)row * ldc + col] = f2bf(v);
        }
      }
    }
  }
}

// ---------------------------------------------------------------------------
// Flash attention: grid (S/128, B*HEADS, 3 combos), 256 threads = 8 waves.
// Wave owns 16 query rows; block streams async double-buffered 32-key tiles:
// K from row-major QKV, V from transposed Vt (keys contiguous -> b128 frags).
// combo: 0 -> (q,k,v), 1 -> (q2,k2,v2), 2 -> (q,k2,v2)
// obf layout: bf16 [combo(3)][b*s(8192)][h*64+d]  (ready for out-proj GEMM)
// ---------------------------------------------------------------------------
__global__ __launch_bounds__(256) void wmma_flash_attn(
    const uint16_t* __restrict__ qkv, const uint16_t* __restrict__ Vt,
    uint16_t* __restrict__ obf) {
  __shared__ uint16_t Ks[2][32][72];     // [buf][key][d], 144B rows
  __shared__ uint16_t Vts[2][64][40];    // [buf][d][key], 80B rows
  __shared__ uint16_t Ps[8][16][32];     // per-wave P tile [m][key]

  const int tid = threadIdx.x, lane = tid & 31, wid = tid >> 5;
  const int half = lane >> 4, ln = lane & 15;
  const int combo = blockIdx.z;
  const int qsrc  = (combo == 1) ? 1 : 0;
  const int kvsrc = (combo == 0) ? 0 : 1;
  const int b = blockIdx.y / 12, h = blockIdx.y % 12;
  const int q0 = blockIdx.x * 128 + wid * 16;

  const uint16_t* Qb = qkv + ((long long)(qsrc * 8192 + b * 1024)) * 2304 + h * 64;
  const uint16_t* Kb = qkv + ((long long)(kvsrc * 8192 + b * 1024)) * 2304 + 768 + h * 64;
  const uint16_t* Vtb = Vt + (((long long)kvsrc * 8 + b) * 12 + h) * 64 * 1024;

  // staging: K tile 32 rows x 8 chunks; V tile 64 rows x 4 chunks (16B chunks)
  const int krow = tid >> 3, kc0 = (tid & 7) * 8;
  const int vrow = tid >> 2, vc0 = (tid & 3) * 8;
  auto stage = [&](int buf, int kb) {
    const uint16_t* ksrc = Kb + (long long)(kb + krow) * 2304 + kc0;
    const uint16_t* vsrc = Vtb + (long long)vrow * 1024 + kb + vc0;
    async_b128(lds_off(&Ks[buf][krow][kc0]), (uint64_t)(uintptr_t)ksrc);
    async_b128(lds_off(&Vts[buf][vrow][vc0]), (uint64_t)(uintptr_t)vsrc);
  };

  // Q fragments (d 0..31 and 32..63), loaded once
  AFrag aq0, aq1;
  {
    const uint16_t* qrow = Qb + (long long)(q0 + ln) * 2304;
#pragma unroll
    for (int j = 0; j < 8; ++j) {
      int k = a_k(j, half);
      aq0.u[j] = *(const uint32_t*)&qrow[k];
      aq1.u[j] = *(const uint32_t*)&qrow[k + 32];
    }
  }

  v8f o[4] = {};
  float mrow[8], rsum[8];
#pragma unroll
  for (int e = 0; e < 8; ++e) { mrow[e] = -3.0e38f; rsum[e] = 0.f; }

  stage(0, 0);
  wait_async0();
  __syncthreads();

  int buf = 0;
  for (int kb = 0; kb < 1024; kb += 32) {
    if (kb + 32 < 1024) stage(buf ^ 1, kb + 32);

    // S = (Q K^T) * scale for two 16-key chunks
    v8f sc[2];
#pragma unroll
    for (int c = 0; c < 2; ++c) {
      AFrag blo, bhi;
      int key = c * 16 + ln;
#pragma unroll
      for (int j = 0; j < 8; ++j) {
        int k = 16 * half + 2 * j;      // k = d index here (contiguous pairs)
        blo.u[j] = *(const uint32_t*)&Ks[buf][key][k];
        bhi.u[j] = *(const uint32_t*)&Ks[buf][key][k + 32];
      }
      v8f s = {};
      s = __builtin_amdgcn_wmma_f32_16x16x32_bf16(false, aq0.v, false, blo.v, (short)0, s, false, false);
      s = __builtin_amdgcn_wmma_f32_16x16x32_bf16(false, aq1.v, false, bhi.v, (short)0, s, false, false);
      sc[c] = s * 0.125f;               // HEAD_DIM^-0.5
    }

    // online softmax over 32 keys; rows live as {e + 8*half} per lane,
    // row reduction = shfl over the 16-lane half-group.
#pragma unroll
    for (int e = 0; e < 8; ++e) {
      float lmax = fmaxf(sc[0][e], sc[1][e]);
#pragma unroll
      for (int off = 1; off < 16; off <<= 1)
        lmax = fmaxf(lmax, __shfl_xor(lmax, off, 32));
      float mnew  = fmaxf(mrow[e], lmax);
      float alpha = __expf(mrow[e] - mnew);
      float p0 = __expf(sc[0][e] - mnew);
      float p1 = __expf(sc[1][e] - mnew);
      Ps[wid][e + 8 * half][ln]      = f2bf(p0);
      Ps[wid][e + 8 * half][16 + ln] = f2bf(p1);
      float ls = p0 + p1;
#pragma unroll
      for (int off = 1; off < 16; off <<= 1)
        ls += __shfl_xor(ls, off, 32);
      rsum[e] = rsum[e] * alpha + ls;
      mrow[e] = mnew;
#pragma unroll
      for (int jf = 0; jf < 4; ++jf) o[jf][e] *= alpha;
    }

    // P (16x32) -> A fragment via per-wave LDS bounce (layout transpose)
    AFrag ap;
#pragma unroll
    for (int j = 0; j < 8; ++j)
      ap.u[j] = *(const uint32_t*)&Ps[wid][ln][a_k(j, half)];

    // O += P @ V, 4 d-chunks of 16; V frag = contiguous key-pairs from Vts
#pragma unroll
    for (int jf = 0; jf < 4; ++jf) {
      AFrag bv;
      int d = jf * 16 + ln;
#pragma unroll
      for (int j = 0; j < 8; ++j)
        bv.u[j] = *(const uint32_t*)&Vts[buf][d][16 * half + 2 * j];
      o[jf] = __builtin_amdgcn_wmma_f32_16x16x32_bf16(
          false, ap.v, false, bv.v, (short)0, o[jf], false, false);
    }

    wait_async0();
    __syncthreads();
    buf ^= 1;
  }

  // normalize and store bf16 O in [b*s][h*64+d] layout
  uint16_t* orow = obf + ((long long)combo * 8192 + b * 1024) * 768 + h * 64;
#pragma unroll
  for (int e = 0; e < 8; ++e) {
    float inv = 1.0f / rsum[e];
    long long s = q0 + e + 8 * half;
#pragma unroll
    for (int jf = 0; jf < 4; ++jf)
      orow[s * 768 + jf * 16 + ln] = f2bf(o[jf][e] * inv);
  }
}

// ---------------------------------------------------------------------------
__global__ void cvt_f32_bf16(const float* __restrict__ src,
                             uint16_t* __restrict__ dst, int n) {
  int i = blockIdx.x * blockDim.x + threadIdx.x;
  int stride = gridDim.x * blockDim.x;
  for (; i < n; i += stride) dst[i] = f2bf(src[i]);
}

// src [R][C] f32 -> dst [C][R] bf16, 32x32 LDS tiles (both sides coalesced)
__global__ __launch_bounds__(256) void cvt_transpose_bf16(
    const float* __restrict__ src, uint16_t* __restrict__ dst, int R, int C) {
  __shared__ uint16_t tile[32][33];
  int c0 = blockIdx.x * 32, r0 = blockIdx.y * 32;
#pragma unroll
  for (int i = 0; i < 4; ++i) {
    int r = r0 + threadIdx.y + i * 8, c = c0 + threadIdx.x;
    tile[threadIdx.y + i * 8][threadIdx.x] = f2bf(src[(long long)r * C + c]);
  }
  __syncthreads();
#pragma unroll
  for (int i = 0; i < 4; ++i) {
    int cc = c0 + threadIdx.y + i * 8, rr = r0 + threadIdx.x;
    dst[(long long)cc * R + rr] = tile[threadIdx.x][threadIdx.y + i * 8];
  }
}

// ---------------------------------------------------------------------------
extern "C" void kernel_launch(void* const* d_in, const int* in_sizes, int n_in,
                              void* d_out, int out_size, void* d_ws, size_t ws_size,
                              hipStream_t stream) {
  const float* x    = (const float*)d_in[0];
  const float* x2   = (const float*)d_in[1];
  const float* Wqkv = (const float*)d_in[2];
  const float* Wout = (const float*)d_in[3];
  const float* bout = (const float*)d_in[4];

  char* ws = (char*)d_ws;
  size_t off = 0;
  auto alloc = [&](size_t bytes) {
    void* p = ws + off;
    off += (bytes + 255) & ~(size_t)255;
    return p;
  };
  uint16_t* Xbf    = (uint16_t*)alloc((size_t)16384 * 768 * 2);   // [x;x2] bf16
  uint16_t* WqkvT  = (uint16_t*)alloc((size_t)2304 * 768 * 2);    // [3H][H]
  uint16_t* WoutT  = (uint16_t*)alloc((size_t)768 * 768 * 2);     // [H][H]
  uint16_t* QKVbf  = (uint16_t*)alloc((size_t)16384 * 2304 * 2);  // Q,K rows
  uint16_t* Vtbf   = (uint16_t*)alloc((size_t)2 * 8 * 12 * 64 * 1024 * 2);
  uint16_t* Obf    = (uint16_t*)alloc((size_t)3 * 8192 * 768 * 2);

  cvt_f32_bf16<<<256, 256, 0, stream>>>(x,  Xbf,                      8192 * 768);
  cvt_f32_bf16<<<256, 256, 0, stream>>>(x2, Xbf + (size_t)8192 * 768, 8192 * 768);
  cvt_transpose_bf16<<<dim3(2304 / 32, 768 / 32), dim3(32, 8), 0, stream>>>(
      Wqkv, WqkvT, 768, 2304);
  cvt_transpose_bf16<<<dim3(768 / 32, 768 / 32), dim3(32, 8), 0, stream>>>(
      Wout, WoutT, 768, 768);

  // QKV = [x;x2] @ Wqkv: Q,K -> row-major QKVbf; V -> transposed Vtbf
  dim3 g1(2304 / 128, 16384 / 128, 1);
  wmma_gemm_bf16<<<g1, 256, 0, stream>>>(Xbf, 768, 0, WqkvT, 768,
                                         QKVbf, nullptr, 0, 2304, nullptr,
                                         Vtbf, 16384, 2304, 768);

  // three fused flash attentions
  dim3 ga(1024 / 128, 8 * 12, 3);
  wmma_flash_attn<<<ga, 256, 0, stream>>>(QKVbf, Vtbf, Obf);

  // y{1,2,3} = O @ Wout + bout  -> f32 into d_out (concatenated)
  dim3 g2(768 / 128, 8192 / 128, 3);
  wmma_gemm_bf16<<<g2, 256, 0, stream>>>(Obf, 768, (long long)8192 * 768,
                                         WoutT, 768,
                                         nullptr, (float*)d_out,
                                         (long long)8192 * 768, 768,
                                         bout, nullptr, 8192, 768, 768);
}
